// TransportNN_2405181685894
// MI455X (gfx1250) — matrix operand order repaired and checked by reading, not compiled, not run on hardware
//
#include <hip/hip_runtime.h>
#include <hip/hip_bf16.h>
#include <math.h>

// ---------------------------------------------------------------------------
// TransportNN forward for MI455X (gfx1250, wave32, WMMA + async-LDS).
// Big GEMMs: v_wmma_f32_16x16x32_f16 with split-f16 (hi+lo) operands
// (Ootomo: hi*hi + hi*lo + lo*hi) for near-f32 accuracy under the T=100
// softmax. Operand tiles are staged into LDS with GLOBAL_LOAD_ASYNC_TO_LDS
// (ASYNCcnt-tracked, double-buffered) and consumed via ds_load_b128.
// Block tile: 32x64 (8 waves, one 16x16 WMMA tile each), K-chunk 64.
// ---------------------------------------------------------------------------

typedef __attribute__((ext_vector_type(16))) _Float16 v16h;
typedef __attribute__((ext_vector_type(8)))  float    v8f;

#define B_  1024
#define NT_ 8192
#define NS_ 8192
#define D_  1024
#define C_  10
#define TEMP_ 100.0f

#define LDS_STR  72          // padded row stride in halfs (bank-conflict free)
#define LDS_AREG 2304        // 32 rows * 72   (one A region: hi or lo)
#define LDS_BREG 4608        // 64 rows * 72   (one B region: hi or lo)
#define LDS_BUF  13824       // 2*LDS_AREG + 2*LDS_BREG halfs per buffer

// ---- WMMA operand loaders (layouts per CDNA5 ISA 7.12.2) -------------------
__device__ __forceinline__ v16h load_a16(const _Float16* p) {
  // A 16x32 f16: halves 0..7 = K[base..base+7], halves 8..15 = K[base+16..+23]
  union { v16h v; uint4 q[2]; } u;
  u.q[0] = *(const uint4*)(p);
  u.q[1] = *(const uint4*)(p + 16);
  return u.v;
}
__device__ __forceinline__ v16h load_b16(const _Float16* p) {
  // B 32x16 f16: halves 0..7 = K[base..base+7], halves 8..15 = K[base+16..+23]
  union { v16h v; uint4 q[2]; } u;
  u.q[0] = *(const uint4*)(p);
  u.q[1] = *(const uint4*)(p + 16);
  return u.v;
}

// One 16-byte async global->LDS transfer per lane (ASYNCcnt-tracked).
__device__ __forceinline__ void async_b128(unsigned int lds_byteoff,
                                           const _Float16* g) {
  asm volatile("global_load_async_to_lds_b128 %0, %1, off"
               :: "v"(lds_byteoff), "v"((unsigned long long)(uintptr_t)g)
               : "memory");
}

// ---------------------------------------------------------------------------
// Core block GEMM: C(32x64) += A[m0..m0+32, :K] * B^T where B is [N,K] rows.
// Returns this wave's 16x16 f32 accumulator (split-f16 corrected).
// ---------------------------------------------------------------------------
__device__ __forceinline__ v8f gemm_core_async(
    const _Float16* __restrict__ Ah, const _Float16* __restrict__ Al,
    const _Float16* __restrict__ Bh, const _Float16* __restrict__ Bl,
    int K, int m0blk, int n0blk) {
  __shared__ _Float16 sm[2 * LDS_BUF];
  const int tid  = threadIdx.x;
  const int lane = tid & 31;
  const int wv   = tid >> 5;
  const int mi   = wv >> 2;          // 0..1  (M sub-tile)
  const int ni   = wv & 3;           // 0..3  (N sub-tile)
  const int lrow = lane & 15;
  const int lhi  = lane >> 4;
  const unsigned int ldsbase = (unsigned int)(uintptr_t)(&sm[0]);

  // issue one K-chunk (64 wide) of A(32 rows) + B(64 rows), hi+lo, into buf
  auto issue = [&](int buf, int k0) {
    #pragma unroll
    for (int pa = 0; pa < 2; ++pa) {            // A: 512 x 16B chunks / 256 thr
      int c   = pa * 256 + tid;
      int reg = c >> 8;                         // 0 = hi, 1 = lo
      int cc  = c & 255;
      int row = cc >> 3;                        // 0..31
      int col = (cc & 7) << 3;                  // 0,8,..,56
      const _Float16* g = (reg ? Al : Ah) + (size_t)(m0blk + row) * K + k0 + col;
      unsigned int l = ldsbase +
          2u * (unsigned)(buf * LDS_BUF + reg * LDS_AREG + row * LDS_STR + col);
      async_b128(l, g);
    }
    #pragma unroll
    for (int pb = 0; pb < 4; ++pb) {            // B: 1024 x 16B chunks
      int c   = pb * 256 + tid;
      int reg = c >> 9;                         // 0 = hi, 1 = lo
      int cc  = c & 511;
      int row = cc >> 3;                        // 0..63
      int col = (cc & 7) << 3;
      const _Float16* g = (reg ? Bl : Bh) + (size_t)(n0blk + row) * K + k0 + col;
      unsigned int l = ldsbase +
          2u * (unsigned)(buf * LDS_BUF + 2 * LDS_AREG + reg * LDS_BREG +
                          row * LDS_STR + col);
      async_b128(l, g);
    }
  };

  v8f a0 = {}; v8f a1 = {}; v8f a2 = {};
  const int nch = K >> 6;
  int cb = 0;
  issue(0, 0);
  for (int ch = 0; ch < nch; ++ch) {
    if (ch + 1 < nch) {                         // prefetch next chunk
      issue(cb ^ 1, (ch + 1) << 6);
      asm volatile("s_wait_asynccnt 6" ::: "memory");   // current chunk done
    } else {
      asm volatile("s_wait_asynccnt 0" ::: "memory");
    }
    __syncthreads();                            // publish LDS to all waves

    const _Float16* sA_h = sm + cb * LDS_BUF + (mi * 16 + lrow) * LDS_STR;
    const _Float16* sA_l = sA_h + LDS_AREG;
    const _Float16* sB_h = sm + cb * LDS_BUF + 2 * LDS_AREG +
                           (ni * 16 + lrow) * LDS_STR;
    const _Float16* sB_l = sB_h + LDS_BREG;
    #pragma unroll
    for (int kk = 0; kk < 64; kk += 32) {
      v16h ah = load_a16(sA_h + kk + lhi * 8);
      v16h al = load_a16(sA_l + kk + lhi * 8);
      v16h bh = load_b16(sB_h + kk + lhi * 8);
      v16h bl = load_b16(sB_l + kk + lhi * 8);
      a0 = __builtin_amdgcn_wmma_f32_16x16x32_f16(false, ah, false, bh, (short)0, a0, false, false);
      a1 = __builtin_amdgcn_wmma_f32_16x16x32_f16(false, ah, false, bl, (short)0, a1, false, false);
      a2 = __builtin_amdgcn_wmma_f32_16x16x32_f16(false, al, false, bh, (short)0, a2, false, false);
    }
    __syncthreads();                            // all waves done with buf cb
    cb ^= 1;
  }
  #pragma unroll
  for (int r = 0; r < 8; ++r) a0[r] += a1[r] + a2[r];
  return a0;
}

// ---- GEMM kernels ----------------------------------------------------------
// Distance GEMM: out[i,j] = -100*(sqrt(max(na[i]+nb[j]-2*dot,1e-12)) + lcost)
// lcost (optional) = sum_c preds[i,c]*m2[c,j]   (ETA = 1)
__global__ __launch_bounds__(256) void kgemm_dist(
    const _Float16* __restrict__ Ah, const _Float16* __restrict__ Al,
    const _Float16* __restrict__ Bh, const _Float16* __restrict__ Bl,
    const float* __restrict__ na, const float* __restrict__ nb,
    const float* __restrict__ preds, const float* __restrict__ m2,
    float* __restrict__ out, int K) {
  const int lane = threadIdx.x & 31;
  const int wv   = threadIdx.x >> 5;
  const int lrow = lane & 15;
  const int lhi  = lane >> 4;
  const int m0blk = blockIdx.y << 5;
  const int n0blk = blockIdx.x << 6;
  v8f acc = gemm_core_async(Ah, Al, Bh, Bl, K, m0blk, n0blk);
  const int j = n0blk + (wv & 3) * 16 + lrow;
  const float nbj = nb[j];
  #pragma unroll
  for (int r = 0; r < 8; ++r) {
    const int i = m0blk + (wv >> 2) * 16 + r + (lhi << 3);
    float sq = na[i] + nbj - 2.0f * acc[r];
    float d  = sqrtf(fmaxf(sq, 1e-12f));
    if (preds) {
      const float* pr = preds + i * C_;
      float lc = 0.0f;
      #pragma unroll
      for (int c = 0; c < C_; ++c) lc += pr[c] * m2[c * NS_ + j];
      d += lc;                                   // ETA = 1
    }
    out[(size_t)i * NS_ + j] = -TEMP_ * d;
  }
}

// Plain GEMM: out[i,j] = dot  (out leading dim N)
__global__ __launch_bounds__(256) void kgemm_plain(
    const _Float16* __restrict__ Ah, const _Float16* __restrict__ Al,
    const _Float16* __restrict__ Bh, const _Float16* __restrict__ Bl,
    float* __restrict__ out, int K, int N) {
  const int lane = threadIdx.x & 31;
  const int wv   = threadIdx.x >> 5;
  const int lrow = lane & 15;
  const int lhi  = lane >> 4;
  const int m0blk = blockIdx.y << 5;
  const int n0blk = blockIdx.x << 6;
  v8f acc = gemm_core_async(Ah, Al, Bh, Bl, K, m0blk, n0blk);
  const int j = n0blk + (wv & 3) * 16 + lrow;
  #pragma unroll
  for (int r = 0; r < 8; ++r)
    out[(size_t)(m0blk + (wv >> 2) * 16 + r + (lhi << 3)) * N + j] = acc[r];
}

// ---- elementwise / reduction helpers ---------------------------------------
__global__ __launch_bounds__(256) void ksplit(const float* __restrict__ src,
                                              _Float16* __restrict__ hi,
                                              _Float16* __restrict__ lo, int n) {
  int idx = blockIdx.x * 256 + threadIdx.x;
  if (idx >= n) return;
  float v = src[idx];
  _Float16 h = (_Float16)v;
  hi[idx] = h;
  lo[idx] = (_Float16)(v - (float)h);
}

// asf [NT, D] f32 -> transposed split [D, NT] f16 hi/lo
__global__ __launch_bounds__(256) void ksplitT(const float* __restrict__ src,
                                               _Float16* __restrict__ hi,
                                               _Float16* __restrict__ lo) {
  int idx = blockIdx.x * 256 + threadIdx.x;       // over D*NT, out-linear
  int d = idx >> 13;                              // / NT (8192)
  int n = idx & (NT_ - 1);
  float v = src[(size_t)n * D_ + d];
  _Float16 h = (_Float16)v;
  hi[idx] = h;
  lo[idx] = (_Float16)(v - (float)h);
}

__global__ __launch_bounds__(256) void knorm(const float* __restrict__ src,
                                             float* __restrict__ out, int cols) {
  __shared__ float red[256];
  const int row = blockIdx.x, t = threadIdx.x;
  const float* r = src + (size_t)row * cols;
  float s = 0.0f;
  for (int j = t; j < cols; j += 256) { float v = r[j]; s += v * v; }
  red[t] = s; __syncthreads();
  for (int w = 128; w > 0; w >>= 1) { if (t < w) red[t] += red[t + w]; __syncthreads(); }
  if (t == 0) out[row] = red[0];
}

__global__ __launch_bounds__(256) void kstats(const float* __restrict__ l,
                                              float* __restrict__ rowm,
                                              float* __restrict__ rowinv, int N) {
  __shared__ float red[256];
  const int row = blockIdx.x, t = threadIdx.x;
  const float* lr = l + (size_t)row * N;
  float m = -3.4e38f;
  for (int j = t; j < N; j += 256) m = fmaxf(m, lr[j]);
  red[t] = m; __syncthreads();
  for (int w = 128; w > 0; w >>= 1) { if (t < w) red[t] = fmaxf(red[t], red[t + w]); __syncthreads(); }
  m = red[0]; __syncthreads();
  float s = 0.0f;
  for (int j = t; j < N; j += 256) s += __expf(lr[j] - m);
  red[t] = s; __syncthreads();
  for (int w = 128; w > 0; w >>= 1) { if (t < w) red[t] += red[t + w]; __syncthreads(); }
  if (t == 0) { rowm[row] = m; rowinv[row] = 1.0f / red[0]; }
}

// logits (in place) -> probs f32, plus f16 hi/lo splits. N fixed at 8192.
__global__ __launch_bounds__(256) void kprob(float* __restrict__ l,
                                             const float* __restrict__ rowm,
                                             const float* __restrict__ rowinv,
                                             _Float16* __restrict__ ph,
                                             _Float16* __restrict__ pl) {
  int idx = blockIdx.x * 256 + threadIdx.x;
  int row = idx >> 13;
  float p = __expf(l[idx] - rowm[row]) * rowinv[row];
  l[idx] = p;
  _Float16 h = (_Float16)p;
  ph[idx] = h;
  pl[idx] = (_Float16)(p - (float)h);
}

__global__ __launch_bounds__(256) void kpreds(const float* __restrict__ xs,
                                              const float* __restrict__ W,
                                              const float* __restrict__ bb,
                                              float* __restrict__ preds) {
  __shared__ float red[256];
  __shared__ float tot[C_];
  const int row = blockIdx.x, t = threadIdx.x;
  float acc[C_] = {};
  const float* xr = xs + (size_t)row * D_;
  for (int d = t; d < D_; d += 256) {
    float xv = xr[d];
    const float* wr = W + d * C_;
    #pragma unroll
    for (int c = 0; c < C_; ++c) acc[c] += xv * wr[c];
  }
  for (int c = 0; c < C_; ++c) {
    red[t] = acc[c]; __syncthreads();
    for (int w = 128; w > 0; w >>= 1) { if (t < w) red[t] += red[t + w]; __syncthreads(); }
    if (t == 0) tot[c] = red[0];
    __syncthreads();
  }
  if (t == 0) {
    float v[C_], m = -3.4e38f;
    #pragma unroll
    for (int c = 0; c < C_; ++c) { v[c] = tot[c] + bb[c]; m = fmaxf(m, v[c]); }
    float s = 0.0f;
    #pragma unroll
    for (int c = 0; c < C_; ++c) s += __expf(v[c] - m);
    float inv = 1.0f / s;
    #pragma unroll
    for (int c = 0; c < C_; ++c) preds[row * C_ + c] = __expf(v[c] - m) * inv;
  }
}

// M2[c,j] = sum_k sld[c,k] * shl[j,k]   -> [C, NS]
__global__ __launch_bounds__(256) void km2(const float* __restrict__ sld,
                                           const float* __restrict__ shl,
                                           float* __restrict__ m2) {
  int j = blockIdx.x * 256 + threadIdx.x;
  if (j >= NS_) return;
  float sv[C_];
  #pragma unroll
  for (int k = 0; k < C_; ++k) sv[k] = shl[j * C_ + k];
  #pragma unroll
  for (int c = 0; c < C_; ++c) {
    float a = 0.0f;
    #pragma unroll
    for (int k = 0; k < C_; ++k) a += sld[c * C_ + k] * sv[k];
    m2[c * NS_ + j] = a;
  }
}

// y[row, c] = sum_k p32[row,k] * atl[k,c]
__global__ __launch_bounds__(256) void ky(const float* __restrict__ p32,
                                          const float* __restrict__ atl,
                                          float* __restrict__ out) {
  __shared__ float red[256];
  __shared__ float tot[C_];
  const int row = blockIdx.x, t = threadIdx.x;
  float acc[C_] = {};
  const float* pr = p32 + (size_t)row * NS_;
  for (int k = t; k < NS_; k += 256) {
    float pv = pr[k];
    const float* ar = atl + k * C_;
    #pragma unroll
    for (int c = 0; c < C_; ++c) acc[c] += pv * ar[c];
  }
  for (int c = 0; c < C_; ++c) {
    red[t] = acc[c]; __syncthreads();
    for (int w = 128; w > 0; w >>= 1) { if (t < w) red[t] += red[t + w]; __syncthreads(); }
    if (t == 0) tot[c] = red[0];
    __syncthreads();
  }
  if (t < C_) out[row * C_ + t] = tot[t];
}

// ---------------------------------------------------------------------------
extern "C" void kernel_launch(void* const* d_in, const int* in_sizes, int n_in,
                              void* d_out, int out_size, void* d_ws, size_t ws_size,
                              hipStream_t stream) {
  (void)in_sizes; (void)n_in; (void)out_size; (void)ws_size;
  const float* x   = (const float*)d_in[0];   // [B, 32, 32]
  const float* tf  = (const float*)d_in[1];   // [NT, 32, 32]
  const float* asf = (const float*)d_in[2];   // [NT, 32, 32]
  const float* sf  = (const float*)d_in[3];   // [NS, 32, 32]
  const float* shl = (const float*)d_in[4];   // [NS, C]
  const float* atl = (const float*)d_in[5];   // [NS, C]
  const float* sld = (const float*)d_in[6];   // [C, C]
  const float* W   = (const float*)d_in[7];   // [D, C]
  const float* bb  = (const float*)d_in[8];   // [C]
  float* out = (float*)d_out;                 // [B, C]

  // workspace carve (256B aligned)
  char* w = (char*)d_ws;
  size_t o = 0;
  auto alloc = [&](size_t bytes) -> void* {
    void* p = w + o;
    o = (o + bytes + 255) & ~(size_t)255;
    return p;
  };
  _Float16* xh  = (_Float16*)alloc((size_t)B_  * D_  * 2);
  _Float16* xl  = (_Float16*)alloc((size_t)B_  * D_  * 2);
  _Float16* th  = (_Float16*)alloc((size_t)NT_ * D_  * 2);  // reused for sf
  _Float16* tl  = (_Float16*)alloc((size_t)NT_ * D_  * 2);  // reused for sf
  _Float16* aTh = (_Float16*)alloc((size_t)D_  * NT_ * 2);
  _Float16* aTl = (_Float16*)alloc((size_t)D_  * NT_ * 2);
  float*    p32 = (float*)   alloc((size_t)B_  * NT_ * 4);  // logits -> probs
  _Float16* ph  = (_Float16*)alloc((size_t)B_  * NT_ * 2);
  _Float16* pl  = (_Float16*)alloc((size_t)B_  * NT_ * 2);
  float*    xs  = (float*)   alloc((size_t)B_  * D_  * 4);
  _Float16* xsh = (_Float16*)alloc((size_t)B_  * D_  * 2);
  _Float16* xsl = (_Float16*)alloc((size_t)B_  * D_  * 2);
  float* nx   = (float*)alloc(B_  * 4);
  float* ntv  = (float*)alloc(NT_ * 4);
  float* nsv  = (float*)alloc(NS_ * 4);
  float* nxs  = (float*)alloc(B_  * 4);
  float* rowm = (float*)alloc(B_  * 4);
  float* rinv = (float*)alloc(B_  * 4);
  float* prd  = (float*)alloc((size_t)B_ * C_ * 4);
  float* m2   = (float*)alloc((size_t)C_ * NS_ * 4);

  const dim3 blk(256);

  // -- prep: splits + norms + label-cost matrix
  ksplit <<<(B_  * D_) / 256, blk, 0, stream>>>(x,  xh, xl, B_ * D_);
  ksplit <<<(NT_ * D_) / 256, blk, 0, stream>>>(tf, th, tl, NT_ * D_);
  ksplitT<<<(NT_ * D_) / 256, blk, 0, stream>>>(asf, aTh, aTl);
  knorm  <<<B_,  blk, 0, stream>>>(x,  nx,  D_);
  knorm  <<<NT_, blk, 0, stream>>>(tf, ntv, D_);
  knorm  <<<NS_, blk, 0, stream>>>(sf, nsv, D_);
  km2    <<<NS_ / 256, blk, 0, stream>>>(sld, shl, m2);

  // -- phase 1: feature transport  (logits -> softmax -> P @ asf)
  kgemm_dist<<<dim3(NT_ / 64, B_ / 32), blk, 0, stream>>>(
      xh, xl, th, tl, nx, ntv, nullptr, nullptr, p32, D_);
  kstats<<<B_, blk, 0, stream>>>(p32, rowm, rinv, NT_);
  kprob <<<(B_ * NT_) / 256, blk, 0, stream>>>(p32, rowm, rinv, ph, pl);
  // tf splits dead now -> reuse th/tl for sf
  ksplit<<<(NS_ * D_) / 256, blk, 0, stream>>>(sf, th, tl, NS_ * D_);
  kgemm_plain<<<dim3(D_ / 64, B_ / 32), blk, 0, stream>>>(
      ph, pl, aTh, aTl, xs, NT_, D_);

  // -- source model head + x_src prep
  knorm <<<B_, blk, 0, stream>>>(xs, nxs, D_);
  ksplit<<<(B_ * D_) / 256, blk, 0, stream>>>(xs, xsh, xsl, B_ * D_);
  kpreds<<<B_, blk, 0, stream>>>(xs, W, bb, prd);

  // -- phase 2: label transport (distance + label cost -> softmax -> @ atl)
  kgemm_dist<<<dim3(NS_ / 64, B_ / 32), blk, 0, stream>>>(
      xsh, xsl, th, tl, nxs, nsv, prd, m2, p32, D_);
  kstats<<<B_, blk, 0, stream>>>(p32, rowm, rinv, NS_);
  kprob <<<(B_ * NS_) / 256, blk, 0, stream>>>(p32, rowm, rinv, ph, pl);
  ky    <<<B_, blk, 0, stream>>>(p32, atl, out);
}
